// GraphAttnMultiHead_24361054503612
// MI455X (gfx1250) — matrix operand order repaired and checked by reading, not compiled, not run on hardware
//
#include <hip/hip_runtime.h>

#define DI __device__ __forceinline__

typedef __attribute__((ext_vector_type(16))) __bf16       v16bf;
typedef __attribute__((ext_vector_type(8)))  float        v8f;
typedef __attribute__((ext_vector_type(4)))  float        fvec4;
typedef __attribute__((ext_vector_type(4)))  unsigned int uvec4;

constexpr int N  = 8192;   // nodes
constexpr int CF = 256;    // IN_F == H*OUT_F
constexpr int H  = 4;      // heads
constexpr int OF = 64;     // out features per head
constexpr float NEG_SLOPE = 0.2f;

DI unsigned pack_bf16(float a, float b) {
  union { __bf16 h[2]; unsigned u; } x;
  x.h[0] = (__bf16)a; x.h[1] = (__bf16)b;
  return x.u;
}

DI v16bf load_bf16x16(const __bf16* p) {  // 16 bf16: p[0..7] and p[16..23], 16B-aligned halves
  union { uvec4 q[2]; v16bf v; } x;
  x.q[0] = *(const uvec4*)p;
  x.q[1] = *(const uvec4*)(p + 16);
  return x.v;
}

DI v8f wmma_bf16(v16bf a, v16bf b, v8f c) {
  // D(16x16,f32) = A(16x32,bf16) x B(32x16,bf16) + C
  return __builtin_amdgcn_wmma_f32_16x16x32_bf16(false, a, false, b, (short)0, c,
                                                 false, false);
}

// -------- shared WMMA GEMM core: C[16x64] = Arows(m0..m0+15) x Bcols(n0..n0+63), K=256
// A is f32 row-major [N, 256]. B element (k,n) lives at B[k*KS + n*NS].
template<int KS, int NS>
DI void mma_core(const float* __restrict__ A, const float* __restrict__ B,
                 int m0, int n0, int r, int half, v8f acc[4]) {
  const int sA = half * 8;    // A-operand K-run base for this lane half
  const int kb = half * 16;   // unused; B reads the same K runs as A (base sA)
  for (int k0 = 0; k0 < CF; k0 += 32) {
    // A tile: lane row = r, K runs {sA..sA+7} and {sA+16..sA+23}
    const float* ap = A + (m0 + r) * CF + k0 + sA;
    fvec4 a0 = *(const fvec4*)(ap);
    fvec4 a1 = *(const fvec4*)(ap + 4);
    fvec4 a2 = *(const fvec4*)(ap + 16);
    fvec4 a3 = *(const fvec4*)(ap + 20);
    v16bf am;
#pragma unroll
    for (int j = 0; j < 4; ++j) {
      am[j]      = (__bf16)a0[j];
      am[4 + j]  = (__bf16)a1[j];
      am[8 + j]  = (__bf16)a2[j];
      am[12 + j] = (__bf16)a3[j];
    }
#pragma unroll
    for (int t = 0; t < 4; ++t) {
      const int n = n0 + t * 16 + r;        // lane = output column
      v16bf bm;
#pragma unroll
      for (int j = 0; j < 16; ++j)
        bm[j] = (__bf16)B[(k0 + sA + (j & 7) + ((j >> 3) << 4)) * KS + n * NS];
      acc[t] = wmma_bf16(am, bm, acc[t]);
    }
  }
}

// -------- kernel 1: support = inputs @ weight (stored transposed, bf16)
//          + f1[h,n] = support·u, f2[h,n] = support·v
__global__ void __launch_bounds__(32)
support_kernel(const float* __restrict__ inputs, const float* __restrict__ weight,
               const float* __restrict__ wu, const float* __restrict__ wv,
               __bf16* __restrict__ supT, float* __restrict__ f1,
               float* __restrict__ f2) {
  const int m0   = blockIdx.x * 16;
  const int h    = blockIdx.y;          // head -> cols h*64 .. h*64+63
  const int lane = threadIdx.x;
  const int r = lane & 15, half = lane >> 4;
  const int sA = half * 8;              // D rows for this lane: sA..sA+7
  v8f acc[4] = {};
  mma_core<CF, 1>(inputs, weight, m0, h * OF, r, half, acc);

  float p1[8] = {}, p2[8] = {};
#pragma unroll
  for (int t = 0; t < 4; ++t) {
    const int c = t * 16 + r;           // column within head
    // supportT[h*64+c][m] in bf16, packed pair stores
    unsigned* sp = (unsigned*)(supT + (size_t)(h * OF + c) * N + m0 + sA);
#pragma unroll
    for (int v = 0; v < 8; v += 2)
      sp[v >> 1] = pack_bf16(acc[t][v], acc[t][v + 1]);
    const float uu = wu[h * OF + c];
    const float vv = wv[h * OF + c];
#pragma unroll
    for (int v = 0; v < 8; ++v) {
      p1[v] = fmaf(acc[t][v], uu, p1[v]);
      p2[v] = fmaf(acc[t][v], vv, p2[v]);
    }
  }
  // reduce over the 16 lanes (columns) sharing the same 8 rows
#pragma unroll
  for (int off = 8; off > 0; off >>= 1)
#pragma unroll
    for (int v = 0; v < 8; ++v) {
      p1[v] += __shfl_xor(p1[v], off, 32);
      p2[v] += __shfl_xor(p2[v], off, 32);
    }
  if (r == 0) {
#pragma unroll
    for (int v = 0; v < 8; ++v) {
      f1[h * N + m0 + sA + v] = p1[v];
      f2[h * N + m0 + sA + v] = p2[v];
    }
  }
}

// -------- kernel 2: base = inputs @ proj_w.T + proj_b + bias  (f32 [N,256])
__global__ void __launch_bounds__(32)
proj_kernel(const float* __restrict__ inputs, const float* __restrict__ projw,
            const float* __restrict__ projb, const float* __restrict__ bias,
            float* __restrict__ base) {
  const int m0   = blockIdx.x * 16;
  const int n0   = blockIdx.y * 64;
  const int lane = threadIdx.x;
  const int r = lane & 15, half = lane >> 4;
  const int sA = half * 8;
  v8f acc[4] = {};
  // (proj_w.T)(k,c) = proj_w[c*256 + k]  ->  KS=1, NS=256 (per-lane contiguous)
  mma_core<1, CF>(inputs, projw, m0, n0, r, half, acc);
#pragma unroll
  for (int t = 0; t < 4; ++t) {
    const int col = n0 + t * 16 + r;
    const float add = projb[col] + bias[col];
#pragma unroll
    for (int v = 0; v < 8; ++v)
      base[(size_t)(m0 + sA + v) * CF + col] = acc[t][v] + add;
  }
}

// -------- kernel 3: fused masked-softmax attention + attn@support + epilogue
// One workgroup = 32 attn rows as two 16-row strips x 4 heads (8 waves).
// Both strips touch identical supportT/f1 cache lines -> WGP$ reuse halves
// L2 traffic on the B operand; adj (HBM-bound, 256 MB) is still read once.
// Softmax computed shift-free (exact: self-loop guarantees denom>0, logits O(20)).
__global__ void __launch_bounds__(256)
attn_kernel(const float* __restrict__ adj, const __bf16* __restrict__ supT,
            const float* __restrict__ f1, const float* __restrict__ f2,
            const float* __restrict__ base, float* __restrict__ out) {
  const int wave  = threadIdx.x >> 5;
  const int h     = wave & 3;                      // head
  const int strip = wave >> 2;                     // 0 or 1
  const int n0    = blockIdx.x * 32 + strip * 16;  // first attn row of this wave
  const int lane  = threadIdx.x & 31;
  const int r = lane & 15, half = lane >> 4;
  const int sA = half * 8;
  const float f2r = f2[h * N + n0 + r];
  const float* adjrow = adj + (size_t)(n0 + r) * N;
  const float* f1h    = f1 + h * N;
  const __bf16* bp    = supT + (size_t)h * OF * N + (size_t)r * N + half * 8;

  v8f acc[4] = {};
  float denom = 0.0f;
  for (int m0 = 0; m0 < N; m0 += 32) {
    const float* ap = adjrow + m0 + sA;
    const float* gp = f1h + m0 + sA;
    __builtin_prefetch(ap + 128, 0, 0);           // global_prefetch_b8 on adj stream
    fvec4 av[4], gv[4];
    av[0] = *(const fvec4*)(ap);      av[1] = *(const fvec4*)(ap + 4);
    av[2] = *(const fvec4*)(ap + 16); av[3] = *(const fvec4*)(ap + 20);
    gv[0] = *(const fvec4*)(gp);      gv[1] = *(const fvec4*)(gp + 4);
    gv[2] = *(const fvec4*)(gp + 16); gv[3] = *(const fvec4*)(gp + 20);
    v16bf am;
    float ds[4] = {0.0f, 0.0f, 0.0f, 0.0f};       // 4 partial sums -> shorter chain
#pragma unroll
    for (int i = 0; i < 4; ++i)
#pragma unroll
      for (int j = 0; j < 4; ++j) {
        const float x = gv[i][j] + f2r;           // logits = f1[m] + f2[n]
        const float l = fmaxf(x, NEG_SLOPE * x);  // leaky_relu
        const float e = av[i][j] * __expf(l);     // adj in {0,1}: multiply == mask
        ds[i] += e;
        am[i * 4 + j] = (__bf16)e;                // bf16: exp can reach ~1e9
      }
    denom += (ds[0] + ds[1]) + (ds[2] + ds[3]);
#pragma unroll
    for (int t = 0; t < 4; ++t) {
      v16bf bm = load_bf16x16(bp + (size_t)(t * 16) * N + m0);
      acc[t] = wmma_bf16(am, bm, acc[t]);
    }
  }
  // combine lane halves (rows r held by lanes r and r+16), then per-D-row recip
  denom += __shfl_xor(denom, 16, 32);
  float rden[8];
#pragma unroll
  for (int v = 0; v < 8; ++v)
    rden[v] = 1.0f / __shfl(denom, sA + v, 32);
#pragma unroll
  for (int t = 0; t < 4; ++t) {
    const int col = h * OF + t * 16 + r;
#pragma unroll
    for (int v = 0; v < 8; ++v) {
      const size_t idx = (size_t)(n0 + sA + v) * CF + col;
      out[idx] = acc[t][v] * rden[v] + base[idx];
    }
  }
}

extern "C" void kernel_launch(void* const* d_in, const int* in_sizes, int n_in,
                              void* d_out, int out_size, void* d_ws, size_t ws_size,
                              hipStream_t stream) {
  (void)in_sizes; (void)n_in; (void)out_size; (void)ws_size;
  const float* inputs = (const float*)d_in[0];
  const float* adj    = (const float*)d_in[1];
  const float* weight = (const float*)d_in[2];
  const float* wu     = (const float*)d_in[3];
  const float* wv     = (const float*)d_in[4];
  const float* bias   = (const float*)d_in[5];
  const float* projw  = (const float*)d_in[6];
  const float* projb  = (const float*)d_in[7];
  float* out = (float*)d_out;

  char* ws = (char*)d_ws;
  float*  base = (float*)ws;                                      // 8 MB f32 [N,256]
  __bf16* supT = (__bf16*)(ws + (size_t)N * CF * sizeof(float));  // 4 MB bf16 [256,N]
  float*  f1   = (float*)(ws + (size_t)N * CF * 6);               // 128 KB [H,N]
  float*  f2   = f1 + H * N;                                      // 128 KB [H,N]

  support_kernel<<<dim3(N / 16, H),       32,  0, stream>>>(inputs, weight, wu, wv,
                                                            supT, f1, f2);
  proj_kernel   <<<dim3(N / 16, CF / 64), 32,  0, stream>>>(inputs, projw, projb,
                                                            bias, base);
  attn_kernel   <<<N / 32,                256, 0, stream>>>(adj, supT, f1, f2,
                                                            base, out);
}
